// CldTextDecoder_53678501265528
// MI455X (gfx1250) — compile-verified, matrix operand order checked by reading, not run on hardware
//
#include <hip/hip_runtime.h>
#include <hip/hip_bf16.h>

// ---------------------------------------------------------------------------
// Model constants (from reference)
// ---------------------------------------------------------------------------
#define BB   256          // batch
#define EE   512          // latent dim
#define PP   512          // lin out
#define DD   768          // model dim
#define HH   8            // heads
#define DHH  96           // head dim
#define CLL  40           // content length
#define PLL  40           // prefix length
#define NN   80           // seq length (CL+PL)
#define LL   8            // layers
#define MHH  1536         // mlp hidden
#define QKV  (3*DD)       // 2304 fused q|k|v width
#define TOK  (BB*NN)      // 20480 tokens
#define SEQ_LD (NN*DD)    // 61440, per-sample row stride of seq buffer
#define EPSF 1e-5f

typedef __bf16 bf16;
typedef bf16  v16bf __attribute__((ext_vector_type(16)));
typedef float v8f   __attribute__((ext_vector_type(8)));
typedef int   v4i   __attribute__((ext_vector_type(4)));

// ---- CDNA5 async-copy path (guarded: falls back to pipelined sync staging) --
#if defined(__gfx1250__) && __has_builtin(__builtin_amdgcn_global_load_async_to_lds_b128)
#define HAVE_ASYNC 1
#define AS1 __attribute__((address_space(1)))
#define AS3 __attribute__((address_space(3)))
__device__ inline void async_b128(const bf16* g, bf16* l) {
  // global -> LDS, 16B per lane, tracked by ASYNCcnt
  __builtin_amdgcn_global_load_async_to_lds_b128(
      (AS1 v4i*)g, (AS3 v4i*)l, 0, 0);
}
#else
#define HAVE_ASYNC 0
#endif

__device__ inline void wait_async_all() {
#if HAVE_ASYNC
#if __has_builtin(__builtin_amdgcn_s_wait_asynccnt)
  __builtin_amdgcn_s_wait_asynccnt(0);
#else
  asm volatile("s_wait_asynccnt 0x0" ::: "memory");
#endif
#endif
}

// ---------------------------------------------------------------------------
// f32 -> bf16 elementwise conversion (for A-side activations: latent)
// ---------------------------------------------------------------------------
__global__ void convert_f32_bf16(const float* __restrict__ in,
                                 bf16* __restrict__ out, size_t n4) {
  size_t i = (size_t)blockIdx.x * blockDim.x + threadIdx.x;
  size_t stride = (size_t)gridDim.x * blockDim.x;
  for (; i < n4; i += stride) {
    float4 f = ((const float4*)in)[i];
    union { bf16 h[4]; uint2 u; } o;
    o.h[0] = (bf16)f.x; o.h[1] = (bf16)f.y;
    o.h[2] = (bf16)f.z; o.h[3] = (bf16)f.w;
    ((uint2*)out)[i] = o.u;
  }
}

// ---------------------------------------------------------------------------
// Transpose + convert: f32 src (K x N, row-major) -> bf16 dst (N x K, row-major)
// 32x32 LDS tiles, 256 threads (32x8). blockIdx.z batches layers.
// N-major weights make GEMM B staging identical to A (pure b128 row copies).
// ---------------------------------------------------------------------------
__global__ __launch_bounds__(256)
void transpose_convert(const float* __restrict__ in, bf16* __restrict__ out,
                       int K, int N, size_t inStrideL, size_t outStrideL) {
  __shared__ float t[32][33];
  const float* src = in  + (size_t)blockIdx.z * inStrideL;
  bf16*        dst = out + (size_t)blockIdx.z * outStrideL;
  const int k0 = blockIdx.y * 32, n0 = blockIdx.x * 32;
  const int tx = threadIdx.x & 31, ty = threadIdx.x >> 5;
#pragma unroll
  for (int i = 0; i < 4; ++i)
    t[ty + i * 8][tx] = src[(size_t)(k0 + ty + i * 8) * N + n0 + tx];
  __syncthreads();
#pragma unroll
  for (int i = 0; i < 4; ++i)
    dst[(size_t)(n0 + ty + i * 8) * K + k0 + tx] = (bf16)t[tx][ty + i * 8];
}

// ---------------------------------------------------------------------------
// WMMA bf16 GEMM:  C(MxN) = A(MxK) * Bt(NxK)^T  [+bias] [+C residual] [relu]
//   A : bf16 row-major, ld = K        Bt: bf16 N-major (pre-transposed), ld = K
//   Cf: f32 out, row stride ldc (if OUT_F32/RES); Cb: bf16 out, row stride N
// Block tile 128(M) x 256(N), K-step 32. 8 waves as 2(M) x 4(N), each wave a
// 64x64 tile = 4x4 WMMAs -> 16 v_wmma per 16 ds_load_b128 (512 B LDS/WMMA).
// LDS double-buffered; next K-tile staged with global_load_async_to_lds_b128
// (ASYNCcnt) while the current tile feeds v_wmma_f32_16x16x32_bf16.
// reuse_a hint (constant) on all but the last WMMA sharing an A fragment.
// ---------------------------------------------------------------------------
union FragU { uint4 q[2]; v16bf v; };

__device__ inline v16bf load_frag(const bf16* rowbase, int hi) {
  FragU f;
  f.q[0] = *(const uint4*)(rowbase + hi * 8);
  f.q[1] = *(const uint4*)(rowbase + 16 + hi * 8);
  return f.v;
}

template <bool BIAS, bool RES, bool RELU, bool OUT_F32, bool OUT_BF16>
__global__ __launch_bounds__(256)
void gemm_bf16(const bf16* __restrict__ A, const bf16* __restrict__ Bt,
               const float* __restrict__ bias,
               float* __restrict__ Cf, bf16* __restrict__ Cb,
               int M, int N, int K, int ldc) {
  __shared__ __align__(16) bf16 As[2][128 * 32];   // 2 x 8 KB
  __shared__ __align__(16) bf16 Bs[2][256 * 32];   // 2 x 16 KB

  const int tid  = threadIdx.x;
  const int lane = tid & 31;
  const int wid  = tid >> 5;
  const int wm   = wid >> 2;      // 0..1 -> rows wm*64
  const int wn   = wid & 3;       // 0..3 -> cols wn*64
  const int hi   = lane >> 4;
  const int lrow = lane & 15;

  const int rowBase = blockIdx.y * 128;
  const int colBase = blockIdx.x * 256;

  v8f acc[4][4] = {};
  const int KT = K >> 5;

  auto stage = [&](int kt, int bufi) {
    const int k0 = kt << 5;
#pragma unroll
    for (int rep = 0; rep < 2; ++rep) {            // A: 512 b128 segments
      int s = tid + rep * 256;
      int row = s >> 2, kc = (s & 3) << 3;
      const bf16* ga = A + (size_t)(rowBase + row) * K + k0 + kc;
#if HAVE_ASYNC
      async_b128(ga, &As[bufi][row * 32 + kc]);
#else
      uint4 d = *(const uint4*)ga;
      *(uint4*)&As[bufi][row * 32 + kc] = d;
#endif
    }
#pragma unroll
    for (int rep = 0; rep < 4; ++rep) {            // B: 1024 b128 segments
      int s = tid + rep * 256;
      int row = s >> 2, kc = (s & 3) << 3;
      const bf16* gb = Bt + (size_t)(colBase + row) * K + k0 + kc;
#if HAVE_ASYNC
      async_b128(gb, &Bs[bufi][row * 32 + kc]);
#else
      uint4 d = *(const uint4*)gb;
      *(uint4*)&Bs[bufi][row * 32 + kc] = d;
#endif
    }
  };

  stage(0, 0);
  wait_async_all();
  __syncthreads();

  for (int kt = 0; kt < KT; ++kt) {
    const int cur = kt & 1;
    if (kt + 1 < KT) stage(kt + 1, cur ^ 1);   // overlap with compute below

    v16bf aF[4], bF[4];
#pragma unroll
    for (int i = 0; i < 4; ++i)
      aF[i] = load_frag(&As[cur][(wm * 64 + i * 16 + lrow) * 32], hi);
#pragma unroll
    for (int j = 0; j < 4; ++j)
      bF[j] = load_frag(&Bs[cur][(wn * 64 + j * 16 + lrow) * 32], hi);

#pragma unroll
    for (int i = 0; i < 4; ++i) {
      acc[i][0] = __builtin_amdgcn_wmma_f32_16x16x32_bf16(
          false, aF[i], false, bF[0], (short)0, acc[i][0], true,  false);
      acc[i][1] = __builtin_amdgcn_wmma_f32_16x16x32_bf16(
          false, aF[i], false, bF[1], (short)0, acc[i][1], true,  false);
      acc[i][2] = __builtin_amdgcn_wmma_f32_16x16x32_bf16(
          false, aF[i], false, bF[2], (short)0, acc[i][2], true,  false);
      acc[i][3] = __builtin_amdgcn_wmma_f32_16x16x32_bf16(
          false, aF[i], false, bF[3], (short)0, acc[i][3], false, false);
    }

    if (kt + 1 < KT) {
      wait_async_all();      // our async writes into buf^1 have landed
      __syncthreads();       // everyone done reading cur / writing next
    }
  }

  // ---- epilogue: lane 0-15 -> M=r, lane 16-31 -> M=8+r, N=lane&15 ----
#pragma unroll
  for (int i = 0; i < 4; ++i) {
#pragma unroll
    for (int j = 0; j < 4; ++j) {
      int rr0 = rowBase + wm * 64 + i * 16 + hi * 8;
      int c   = colBase + wn * 64 + j * 16 + lrow;
#pragma unroll
      for (int r = 0; r < 8; ++r) {
        int rr = rr0 + r;
        float v = acc[i][j][r];
        if (BIAS) v += bias[c];
        if (RES)  v += Cf[(size_t)rr * ldc + c];
        if (RELU) v = fmaxf(v, 0.0f);
        if (OUT_F32)  Cf[(size_t)rr * ldc + c] = v;
        if (OUT_BF16) Cb[(size_t)rr * N + c] = (bf16)v;
      }
    }
  }
}

// ---------------------------------------------------------------------------
// Prefix fill: seq[b][CL+p][d] = prefix_const[p][d]
// ---------------------------------------------------------------------------
__global__ void fill_prefix(const float* __restrict__ prefix,
                            float* __restrict__ seq) {
  int idx = blockIdx.x * blockDim.x + threadIdx.x;
  if (idx >= BB * PLL * DD) return;
  int b = idx / (PLL * DD);
  int r = idx % (PLL * DD);
  seq[(size_t)b * SEQ_LD + CLL * DD + r] = prefix[r];
}

// ---------------------------------------------------------------------------
// LayerNorm over D=768: one block per token -> bf16 out (GEMM input)
// ---------------------------------------------------------------------------
__global__ __launch_bounds__(256)
void layernorm_bf16(const float* __restrict__ x, const float* __restrict__ sc,
                    const float* __restrict__ sh, bf16* __restrict__ out) {
  __shared__ float rs[256], rq[256];
  const int t   = blockIdx.x;
  const int tid = threadIdx.x;
  const float* xr = x + (size_t)t * DD;

  float x0 = xr[tid], x1 = xr[tid + 256], x2 = xr[tid + 512];
  rs[tid] = x0 + x1 + x2;
  rq[tid] = x0 * x0 + x1 * x1 + x2 * x2;
  __syncthreads();
#pragma unroll
  for (int off = 128; off > 0; off >>= 1) {
    if (tid < off) { rs[tid] += rs[tid + off]; rq[tid] += rq[tid + off]; }
    __syncthreads();
  }
  const float mean = rs[0] * (1.0f / DD);
  const float var  = rq[0] * (1.0f / DD) - mean * mean;
  const float inv  = rsqrtf(var + EPSF);

  bf16* orow = out + (size_t)t * DD;
  orow[tid]       = (bf16)((x0 - mean) * inv * sc[tid]       + sh[tid]);
  orow[tid + 256] = (bf16)((x1 - mean) * inv * sc[tid + 256] + sh[tid + 256]);
  orow[tid + 512] = (bf16)((x2 - mean) * inv * sc[tid + 512] + sh[tid + 512]);
}

// ---------------------------------------------------------------------------
// Attention: one block per (batch, head). N=80, DH=96, ~5 GF total -> VALU.
// Reads fused qkv (bf16, row stride 2304): q | k(+768) | v(+1536).
// ---------------------------------------------------------------------------
__global__ __launch_bounds__(128)
void attention_kernel(const bf16* __restrict__ qkv, bf16* __restrict__ o) {
  __shared__ bf16  Ksh[NN * DHH];
  __shared__ bf16  Vsh[NN * DHH];
  __shared__ float Psh[NN * NN];

  const int b = blockIdx.x / HH;
  const int h = blockIdx.x % HH;
  const int tid = threadIdx.x;
  const float scale = 0.102062072615966f;   // 96^-0.5

  const bf16* kbase = qkv + (size_t)b * NN * QKV + DD + h * DHH;
  const bf16* vbase = kbase + DD;
  for (int i = tid; i < NN * DHH; i += 128) {
    int m = i / DHH, d = i % DHH;
    Ksh[i] = kbase[(size_t)m * QKV + d];
    Vsh[i] = vbase[(size_t)m * QKV + d];
  }
  __syncthreads();

  if (tid < NN) {
    const int n = tid;
    float qr[DHH];
    const bf16* qrow = qkv + ((size_t)b * NN + n) * QKV + h * DHH;
#pragma unroll
    for (int d = 0; d < DHH; ++d) qr[d] = (float)qrow[d];

    float mx = -3.0e38f;
    for (int m = 0; m < NN; ++m) {
      float s = 0.f;
      const bf16* kr = &Ksh[m * DHH];
#pragma unroll
      for (int d = 0; d < DHH; ++d) s += qr[d] * (float)kr[d];
      s *= scale;
      Psh[n * NN + m] = s;
      mx = fmaxf(mx, s);
    }
    float sum = 0.f;
    for (int m = 0; m < NN; ++m) {
      float e = __expf(Psh[n * NN + m] - mx);
      Psh[n * NN + m] = e;
      sum += e;
    }
    const float rcp = 1.0f / sum;
    bf16* orow = o + ((size_t)b * NN + n) * DD + h * DHH;
    for (int d = 0; d < DHH; ++d) {
      float a = 0.f;
      for (int m = 0; m < NN; ++m) a += Psh[n * NN + m] * (float)Vsh[m * DHH + d];
      orow[d] = (bf16)(a * rcp);
    }
  }
}

// ---------------------------------------------------------------------------
// Output slice: out[b][p][d] = seq[b][CL+p][d]
// ---------------------------------------------------------------------------
__global__ void copy_out(const float* __restrict__ seq, float* __restrict__ out) {
  int idx = blockIdx.x * blockDim.x + threadIdx.x;
  if (idx >= BB * PLL * DD) return;
  int b = idx / (PLL * DD);
  int r = idx % (PLL * DD);
  out[idx] = seq[(size_t)b * SEQ_LD + CLL * DD + r];
}

// ---------------------------------------------------------------------------
// Host-side orchestration
// ---------------------------------------------------------------------------
extern "C" void kernel_launch(void* const* d_in, const int* in_sizes, int n_in,
                              void* d_out, int out_size, void* d_ws, size_t ws_size,
                              hipStream_t stream) {
  const float* latent = (const float*)d_in[0];
  const float* lin_w  = (const float*)d_in[1];
  const float* lin_b  = (const float*)d_in[2];
  const float* map_w  = (const float*)d_in[3];
  const float* map_b  = (const float*)d_in[4];
  const float* prefix = (const float*)d_in[5];
  const float* ln1_s  = (const float*)d_in[6];
  const float* ln1_b  = (const float*)d_in[7];
  const float* wq     = (const float*)d_in[8];
  const float* wkv    = (const float*)d_in[9];
  const float* wo     = (const float*)d_in[10];
  const float* bo     = (const float*)d_in[11];
  const float* ln2_s  = (const float*)d_in[12];
  const float* ln2_b  = (const float*)d_in[13];
  const float* w1     = (const float*)d_in[14];
  const float* b1     = (const float*)d_in[15];
  const float* w2     = (const float*)d_in[16];
  const float* b2     = (const float*)d_in[17];

  char* p = (char*)d_ws;
  auto carve = [&](size_t bytes) -> void* {
    void* r = (void*)p;
    p += (bytes + 255) & ~(size_t)255;
    return r;
  };

  // N-major (transposed) bf16 weights
  bf16* lin_w_t = (bf16*)carve((size_t)PP * EE * 2);
  bf16* map_w_t = (bf16*)carve((size_t)CLL * DD * PP * 2);
  bf16* wqkv_t  = (bf16*)carve((size_t)LL * QKV * DD * 2);
  bf16* wo_t    = (bf16*)carve((size_t)LL * DD * DD * 2);
  bf16* w1_t    = (bf16*)carve((size_t)LL * MHH * DD * 2);
  bf16* w2_t    = (bf16*)carve((size_t)LL * DD * MHH * 2);

  // activations
  bf16*  lat_in = (bf16*)carve((size_t)BB * EE * 2);
  bf16*  lat_b  = (bf16*)carve((size_t)BB * PP * 2);
  float* seq    = (float*)carve((size_t)BB * SEQ_LD * 4);
  bf16*  h_b    = (bf16*)carve((size_t)TOK * DD * 2);
  bf16*  qkv_b  = (bf16*)carve((size_t)TOK * QKV * 2);
  bf16*  o_b    = (bf16*)carve((size_t)TOK * DD * 2);
  bf16*  m1_b   = (bf16*)carve((size_t)TOK * MHH * 2);

  // ---- weights: transpose + convert to bf16 (N-major) ----
  dim3 tb(256);
  transpose_convert<<<dim3(PP / 32, EE / 32, 1), tb, 0, stream>>>(
      lin_w, lin_w_t, EE, PP, 0, 0);
  transpose_convert<<<dim3((CLL * DD) / 32, PP / 32, 1), tb, 0, stream>>>(
      map_w, map_w_t, PP, CLL * DD, 0, 0);
  transpose_convert<<<dim3(DD / 32, DD / 32, LL), tb, 0, stream>>>(
      wq, wqkv_t, DD, DD, (size_t)DD * DD, (size_t)QKV * DD);
  transpose_convert<<<dim3((2 * DD) / 32, DD / 32, LL), tb, 0, stream>>>(
      wkv, wqkv_t + (size_t)DD * DD, DD, 2 * DD, (size_t)DD * 2 * DD, (size_t)QKV * DD);
  transpose_convert<<<dim3(DD / 32, DD / 32, LL), tb, 0, stream>>>(
      wo, wo_t, DD, DD, (size_t)DD * DD, (size_t)DD * DD);
  transpose_convert<<<dim3(MHH / 32, DD / 32, LL), tb, 0, stream>>>(
      w1, w1_t, DD, MHH, (size_t)DD * MHH, (size_t)MHH * DD);
  transpose_convert<<<dim3(DD / 32, MHH / 32, LL), tb, 0, stream>>>(
      w2, w2_t, MHH, DD, (size_t)MHH * DD, (size_t)DD * MHH);

  { // latent -> bf16 (A side, no transpose)
    size_t n4 = ((size_t)BB * EE) >> 2;
    convert_f32_bf16<<<(int)((n4 + 255) / 256), 256, 0, stream>>>(latent, lat_in, n4);
  }

  // ---- lat = latent @ lin_w + lin_b (bf16 out) ----
  gemm_bf16<true, false, false, false, true><<<dim3(PP / 256, BB / 128), 256, 0, stream>>>(
      lat_in, lin_w_t, lin_b, nullptr, lat_b, BB, PP, EE, PP);

  // ---- x = lat @ map_w + map_b, straight into seq[:, :CL, :] via ldc ----
  gemm_bf16<true, false, false, true, false><<<dim3((CLL * DD) / 256, BB / 128), 256, 0, stream>>>(
      lat_b, map_w_t, map_b, seq, nullptr, BB, CLL * DD, PP, SEQ_LD);

  // ---- seq[:, CL:, :] = prefix ----
  fill_prefix<<<(BB * PLL * DD + 255) / 256, 256, 0, stream>>>(prefix, seq);

  // ---- transformer layers ----
  for (int l = 0; l < LL; ++l) {
    const bf16* wqkv_l = wqkv_t + (size_t)l * QKV * DD;
    const bf16* wo_l   = wo_t   + (size_t)l * DD * DD;
    const bf16* w1_l   = w1_t   + (size_t)l * MHH * DD;
    const bf16* w2_l   = w2_t   + (size_t)l * DD * MHH;

    layernorm_bf16<<<TOK, 256, 0, stream>>>(seq, ln1_s + l * DD, ln1_b + l * DD, h_b);

    // fused qkv = h @ [wq|wkv]
    gemm_bf16<false, false, false, false, true><<<dim3(QKV / 256, TOK / 128), 256, 0, stream>>>(
        h_b, wqkv_l, nullptr, nullptr, qkv_b, TOK, QKV, DD, QKV);

    attention_kernel<<<BB * HH, 128, 0, stream>>>(qkv_b, o_b);

    // seq += o @ wo + bo
    gemm_bf16<true, true, false, true, false><<<dim3(DD / 256, TOK / 128), 256, 0, stream>>>(
        o_b, wo_l, bo + l * DD, seq, nullptr, TOK, DD, DD, DD);

    layernorm_bf16<<<TOK, 256, 0, stream>>>(seq, ln2_s + l * DD, ln2_b + l * DD, h_b);

    // m1 = relu(h @ w1 + b1)
    gemm_bf16<true, false, true, false, true><<<dim3(MHH / 256, TOK / 128), 256, 0, stream>>>(
        h_b, w1_l, b1 + (size_t)l * MHH, nullptr, m1_b, TOK, MHH, DD, MHH);

    // seq += m1 @ w2 + b2
    gemm_bf16<true, true, false, true, false><<<dim3(DD / 256, TOK / 128), 256, 0, stream>>>(
        m1_b, w2_l, b2 + l * DD, seq, nullptr, TOK, DD, MHH, DD);
  }

  copy_out<<<(BB * PLL * DD + 255) / 256, 256, 0, stream>>>(seq, (float*)d_out);

  (void)in_sizes; (void)n_in; (void)out_size; (void)ws_size;
}
